// PrototypicalClassifier_90108413870193
// MI455X (gfx1250) — matrix operand order repaired and checked by reading, not compiled, not run on hardware
//
#include <hip/hip_runtime.h>
#include <hip/hip_bf16.h>
#include <math.h>

// ---------------------------------------------------------------------------
// Shapes: M=2048, C=16, K=8, D=256, H1=256, H2=128, CK=C*K=128
// ---------------------------------------------------------------------------
#define M_DIM 2048
#define C_DIM 16
#define K_DIM 8
#define D_DIM 256
#define H1_DIM 256
#define H2_DIM 128
#define CK_DIM 128

typedef __attribute__((ext_vector_type(16))) __bf16 v16bf;
typedef __attribute__((ext_vector_type(8)))  float  v8f;

__device__ __forceinline__ v8f wmma_bf16(v16bf a, v16bf b, v8f c) {
    // (neg_a, A, neg_b, B, c_mod, C, reuse_a, reuse_b)
    return __builtin_amdgcn_wmma_f32_16x16x32_bf16(false, a, false, b,
                                                   (short)0, c, false, false);
}

// ---------------------------------------------------------------------------
// Kernel 1: build bf16 transposed weight copies.
//   WqT[n][k] = W1[k][n]          (256x256)
//   WpT[n][k] = W1[256+k][n]      (256x256)
//   W2T[n][k] = W2[k][n]          (128x256)
// ---------------------------------------------------------------------------
__global__ void convert_weights(const float* __restrict__ W1,
                                const float* __restrict__ W2,
                                __bf16* __restrict__ WqT,
                                __bf16* __restrict__ WpT,
                                __bf16* __restrict__ W2T) {
    int idx = blockIdx.x * blockDim.x + threadIdx.x;
    if (idx < 65536) {
        int n = idx >> 8, k = idx & 255;
        WqT[idx] = (__bf16)W1[k * H1_DIM + n];
    } else if (idx < 131072) {
        int t = idx - 65536;
        int n = t >> 8, k = t & 255;
        WpT[t] = (__bf16)W1[(D_DIM + k) * H1_DIM + n];
    } else if (idx < 163840) {
        int t = idx - 131072;
        int n = t >> 8, k = t & 255;
        W2T[t] = (__bf16)W2[k * H2_DIM + n];
    }
}

// ---------------------------------------------------------------------------
// Kernel 2: C[rows x 256] = A(f32, rows x 256) @ B  via bf16 WMMA, f32 accum.
// BT is the bf16 transposed weight (256 rows of 256 K-values each).
// One 16x16 output tile per wave; K loop = 8 steps of 32.
// Optional per-column bias (b1 for the prototype path).
// ---------------------------------------------------------------------------
__global__ __launch_bounds__(128) void gemm_bf16_256(
        const float* __restrict__ A, const __bf16* __restrict__ BT,
        const float* __restrict__ bias, float* __restrict__ Cout,
        int totalTiles) {
    const int wave = threadIdx.x >> 5;
    const int l    = threadIdx.x & 31;
    const int tile = blockIdx.x * (blockDim.x >> 5) + wave;
    if (tile >= totalTiles) return;           // wave-uniform guard (EXEC stays full)

    const int colTiles = 16;                  // 256 columns / 16
    const int rt = tile / colTiles, ct = tile % colTiles;
    const int row0 = rt * 16, col0 = ct * 16;
    const int hl = l >> 4, ln = l & 15;

    const float* arow = A + (size_t)(row0 + ln) * 256;
    v8f acc = {0.f, 0.f, 0.f, 0.f, 0.f, 0.f, 0.f, 0.f};

    for (int ks = 0; ks < 8; ++ks) {
        const int k0 = ks * 32;
        // A operand: lane(row=ln) holds 16 contiguous K at k0+hl*16
        const float4 f0 = *(const float4*)(arow + k0 + hl * 16);
        const float4 f1 = *(const float4*)(arow + k0 + hl * 16 + 4);
        const float4 f2 = *(const float4*)(arow + k0 + hl * 16 + 8);
        const float4 f3 = *(const float4*)(arow + k0 + hl * 16 + 12);
        v16bf a;
        a[0]  = (__bf16)f0.x;  a[1]  = (__bf16)f0.y;
        a[2]  = (__bf16)f0.z;  a[3]  = (__bf16)f0.w;
        a[4]  = (__bf16)f1.x;  a[5]  = (__bf16)f1.y;
        a[6]  = (__bf16)f1.z;  a[7]  = (__bf16)f1.w;
        a[8]  = (__bf16)f2.x;  a[9]  = (__bf16)f2.y;
        a[10] = (__bf16)f2.z;  a[11] = (__bf16)f2.w;
        a[12] = (__bf16)f3.x;  a[13] = (__bf16)f3.y;
        a[14] = (__bf16)f3.z;  a[15] = (__bf16)f3.w;
        // B operand: lane(col=ln) holds 16 contiguous K at k0+hl*16
        const v16bf b = *(const v16bf*)(BT + (size_t)(col0 + ln) * 256 + k0 + hl * 16);
        acc = wmma_bf16(a, b, acc);
    }

    const float bi = bias ? bias[col0 + ln] : 0.f;
#pragma unroll
    for (int j = 0; j < 8; ++j) {
        // C layout: VGPR j -> row j + 8*hl, column ln
        Cout[(size_t)(row0 + j + 8 * hl) * 256 + col0 + ln] = acc[j] + bi;
    }
}

// ---------------------------------------------------------------------------
// Kernel 3: fused per-query-row pipeline. One block (8 waves) per m.
//   A2[ck,:] = relu(qh[m,:] + phb[ck,:])  (phb already includes b1)
//   h2 = relu(A2 @ W2 + b2); score[ck] = h2 . W3 + b3   -> LDS
//   softmax over K=8 per c; weighted prototype sum; -||q - w||.
// Wave w owns ck rows [16w, 16w+16); 8 accumulators cover all 128 H2 cols.
// ---------------------------------------------------------------------------
__global__ __launch_bounds__(256) void fused_scores_distance(
        const float* __restrict__ qh, const float* __restrict__ phb,
        const __bf16* __restrict__ W2T,
        const float* __restrict__ b2, const float* __restrict__ W3,
        const float* __restrict__ b3,
        const float* __restrict__ query, const float* __restrict__ protos,
        float* __restrict__ out) {
    const int m = blockIdx.x;
    const int w = threadIdx.x >> 5;
    const int l = threadIdx.x & 31;
    const int hl = l >> 4, ln = l & 15;

    __shared__ float s_scores[CK_DIM];

    const float* qrow = qh + (size_t)m * H1_DIM;
    const int ck = w * 16 + ln;                    // this lane's A-row
    const float* prow = phb + (size_t)ck * H1_DIM;

    v8f acc[8];
#pragma unroll
    for (int nt = 0; nt < 8; ++nt)
        acc[nt] = (v8f){0.f, 0.f, 0.f, 0.f, 0.f, 0.f, 0.f, 0.f};

    for (int ks = 0; ks < 8; ++ks) {
        const int k0 = ks * 32;
        const int ka = k0 + hl * 16;
        const int kb = k0 + hl * 16 + 8;
        const float4 q0 = *(const float4*)(qrow + ka);
        const float4 q1 = *(const float4*)(qrow + ka + 4);
        const float4 q2 = *(const float4*)(qrow + kb);
        const float4 q3 = *(const float4*)(qrow + kb + 4);
        const float4 p0 = *(const float4*)(prow + ka);
        const float4 p1 = *(const float4*)(prow + ka + 4);
        const float4 p2 = *(const float4*)(prow + kb);
        const float4 p3 = *(const float4*)(prow + kb + 4);
        v16bf a;
        a[0]  = (__bf16)fmaxf(q0.x + p0.x, 0.f);
        a[1]  = (__bf16)fmaxf(q0.y + p0.y, 0.f);
        a[2]  = (__bf16)fmaxf(q0.z + p0.z, 0.f);
        a[3]  = (__bf16)fmaxf(q0.w + p0.w, 0.f);
        a[4]  = (__bf16)fmaxf(q1.x + p1.x, 0.f);
        a[5]  = (__bf16)fmaxf(q1.y + p1.y, 0.f);
        a[6]  = (__bf16)fmaxf(q1.z + p1.z, 0.f);
        a[7]  = (__bf16)fmaxf(q1.w + p1.w, 0.f);
        a[8]  = (__bf16)fmaxf(q2.x + p2.x, 0.f);
        a[9]  = (__bf16)fmaxf(q2.y + p2.y, 0.f);
        a[10] = (__bf16)fmaxf(q2.z + p2.z, 0.f);
        a[11] = (__bf16)fmaxf(q2.w + p2.w, 0.f);
        a[12] = (__bf16)fmaxf(q3.x + p3.x, 0.f);
        a[13] = (__bf16)fmaxf(q3.y + p3.y, 0.f);
        a[14] = (__bf16)fmaxf(q3.z + p3.z, 0.f);
        a[15] = (__bf16)fmaxf(q3.w + p3.w, 0.f);
#pragma unroll
        for (int nt = 0; nt < 8; ++nt) {
            const v16bf b = *(const v16bf*)(W2T + (size_t)(nt * 16 + ln) * H1_DIM
                                                 + k0 + hl * 16);
            acc[nt] = wmma_bf16(a, b, acc[nt]);
        }
    }

    // Epilogue: +b2, relu, project onto W3; C layout row = j + 8*hl, col = ln.
    float s[8] = {0.f, 0.f, 0.f, 0.f, 0.f, 0.f, 0.f, 0.f};
#pragma unroll
    for (int nt = 0; nt < 8; ++nt) {
        const int ncol = nt * 16 + ln;
        const float bb = b2[ncol];
        const float wv = W3[ncol];
#pragma unroll
        for (int j = 0; j < 8; ++j) {
            float h = acc[nt][j] + bb;
            h = h > 0.f ? h : 0.f;
            s[j] += h * wv;
        }
    }
    // Reduce over the 16 lanes of each half (columns of the tile).
#pragma unroll
    for (int mask = 1; mask <= 8; mask <<= 1)
#pragma unroll
        for (int j = 0; j < 8; ++j) s[j] += __shfl_xor(s[j], mask);

    if (ln == 0) {
        const float b3v = b3[0];
#pragma unroll
        for (int j = 0; j < 8; ++j)
            s_scores[w * 16 + 8 * hl + j] = s[j] + b3v;
    }
    __syncthreads();

    // ---- softmax over K=8 + weighted prototype distance; 2 classes per wave.
    const float* qf = query + (size_t)m * D_DIM;
#pragma unroll
    for (int cc = 0; cc < 2; ++cc) {
        const int c = w * 2 + cc;
        float sc[K_DIM];
        float mx = -3.4e38f;
#pragma unroll
        for (int k = 0; k < K_DIM; ++k) {
            sc[k] = s_scores[c * K_DIM + k];
            mx = fmaxf(mx, sc[k]);
        }
        float den = 0.f;
#pragma unroll
        for (int k = 0; k < K_DIM; ++k) { sc[k] = __expf(sc[k] - mx); den += sc[k]; }
        const float inv = 1.f / den;

        float dacc = 0.f;
#pragma unroll
        for (int i = 0; i < 8; ++i) {
            const int d = l + 32 * i;
            float ws = 0.f;
#pragma unroll
            for (int k = 0; k < K_DIM; ++k)
                ws += sc[k] * inv * protos[(size_t)(c * K_DIM + k) * D_DIM + d];
            const float diff = qf[d] - ws;
            dacc += diff * diff;
        }
#pragma unroll
        for (int mask = 16; mask >= 1; mask >>= 1) dacc += __shfl_xor(dacc, mask);
        if (l == 0) out[m * C_DIM + c] = -sqrtf(dacc);
    }
}

// ---------------------------------------------------------------------------
// Launch
// ---------------------------------------------------------------------------
extern "C" void kernel_launch(void* const* d_in, const int* in_sizes, int n_in,
                              void* d_out, int out_size, void* d_ws, size_t ws_size,
                              hipStream_t stream) {
    (void)in_sizes; (void)n_in; (void)out_size; (void)ws_size;
    const float* query  = (const float*)d_in[0];   // 2048 x 256
    const float* protos = (const float*)d_in[1];   // 16 x 8 x 256
    const float* W1     = (const float*)d_in[2];   // 512 x 256
    const float* b1     = (const float*)d_in[3];   // 256
    const float* W2     = (const float*)d_in[4];   // 256 x 128
    const float* b2     = (const float*)d_in[5];   // 128
    const float* W3     = (const float*)d_in[6];   // 128 x 1
    const float* b3     = (const float*)d_in[7];   // 1
    float* out = (float*)d_out;                    // 2048 x 16

    // Workspace layout (bytes):
    char* ws = (char*)d_ws;
    __bf16* WqT = (__bf16*)(ws + 0);              // 256*256*2 = 131072
    __bf16* WpT = (__bf16*)(ws + 131072);         // 131072
    __bf16* W2T = (__bf16*)(ws + 262144);         // 128*256*2 = 65536
    float*  qh  = (float*)(ws + 327680);          // 2048*256*4 = 2097152
    float*  phb = (float*)(ws + 2424832);         // 128*256*4  = 131072
    // total 2555904 bytes

    // 1. bf16 transposed weights
    convert_weights<<<640, 256, 0, stream>>>(W1, W2, WqT, WpT, W2T);

    // 2. qh = Q @ Wq               (2048x256 tiles -> 2048 wave-tiles, 4/block)
    gemm_bf16_256<<<512, 128, 0, stream>>>(query, WqT, nullptr, qh, 2048);

    // 3. phb = P @ Wp + b1         (128x256 -> 128 wave-tiles)
    gemm_bf16_256<<<32, 128, 0, stream>>>(protos, WpT, b1, phb, 128);

    // 4. fused: relu-add, GEMM2 (WMMA), +b2/relu/W3, softmax, distance
    fused_scores_distance<<<M_DIM, 256, 0, stream>>>(
        qh, phb, W2T, b2, W3, b3, query, protos, out);
}
